// WorldModel_9380208574932
// MI455X (gfx1250) — compile-verified, matrix operand order checked by reading, not run on hardware
//
#include <hip/hip_runtime.h>
#include <hip/hip_bf16.h>

typedef __bf16 bf16;
typedef __attribute__((ext_vector_type(16))) __bf16 v16bf;
typedef __attribute__((ext_vector_type(8)))  float  v8f;
typedef __attribute__((ext_vector_type(4)))  unsigned int u32x4;
typedef __attribute__((address_space(3))) bf16  lbf16;   // explicit LDS pointers
typedef __attribute__((address_space(3))) u32x4 lu32x4;  // -> ds_load/store_b128

// ---------------------------------------------------------------------------
// CDNA5 v_wmma_f32_16x16x32_bf16, wave32.
// D = A*B with A = WT tile (16 out-cols x 32 K), B = X^T tile (32 K x 16 rows).
// Both fragments read 2 x 16 contiguous bytes per lane from row-major
// WT[N x K] / X[M x K]:
//   lane L: row = L&15, kb = (L>>4)*8; f[0..7]=row[k+kb..], f[8..15]=row[k+kb+16..]
// D: lane L, VGPR i -> out col c + (L>>4)*8 + i, batch row r0 + (L&15)
//   => 8 consecutive out cols per lane: one packed 16B bf16 store.
// ---------------------------------------------------------------------------
__device__ inline v16bf load_frag(const bf16* __restrict__ mat, int ld, int k,
                                  int lane) {
  const bf16* p = mat + (size_t)(lane & 15) * ld + k + ((lane >> 4) << 3);
  union { u32x4 q[2]; v16bf f; } u;
  u.q[0] = *(const u32x4*)p;
  u.q[1] = *(const u32x4*)(p + 16);
  return u.f;
}
__device__ inline v16bf load_frag(const lbf16* mat, int ld, int k, int lane) {
  const lbf16* p = mat + (lane & 15) * ld + k + ((lane >> 4) << 3);
  union { u32x4 q[2]; v16bf f; } u;
  u.q[0] = *(const lu32x4*)p;
  u.q[1] = *(const lu32x4*)(p + 16);
  return u.f;
}

__device__ inline v8f wmma_bf16(v16bf a, v16bf b, v8f c) {
  return __builtin_amdgcn_wmma_f32_16x16x32_bf16(false, a, false, b, (short)0, c,
                                                 false, false);
}

// dual N-tile inner loop: one B fragment feeds two WMMAs (3 loads / wmma)
template <typename BP>
__device__ inline void gemm_core(BP Bx, int bld, const bf16* __restrict__ Aw0,
                                 const bf16* __restrict__ Aw1, int ldk, int K,
                                 int lane, v8f& acc0, v8f& acc1) {
  for (int k = 0; k < K; k += 32) {
    v16bf b = load_frag(Bx, bld, k, lane);
    acc0 = wmma_bf16(load_frag(Aw0, ldk, k, lane), b, acc0);
    acc1 = wmma_bf16(load_frag(Aw1, ldk, k, lane), b, acc1);
  }
}

__device__ inline float wred_max(float v) {
#pragma unroll
  for (int m = 16; m > 0; m >>= 1) v = fmaxf(v, __shfl_xor(v, m, 32));
  return v;
}
__device__ inline float wred_sum(float v) {
#pragma unroll
  for (int m = 16; m > 0; m >>= 1) v += __shfl_xor(v, m, 32);
  return v;
}

// ---------------------------------------------------------------------------
// Generic WMMA GEMM: out[256 x N] = act( X[256 x K] * W[K x N] + bias ),
// weights pre-transposed WT[N x K]. Block = 8 waves, tile 16 rows x 256 cols;
// each wave owns two adjacent 16x16 D tiles. X tile staged in LDS (b128).
// ---------------------------------------------------------------------------
__global__ void __launch_bounds__(256) k_gemm(
    const bf16* __restrict__ X, int ldx, const bf16* __restrict__ WT, int ldk,
    const float* __restrict__ bias, int K, int N,
    bf16* __restrict__ outb, int ldob, float* __restrict__ outf, int ldof,
    int act, int use_lds) {
  extern __shared__ char dynsmem[];
  lbf16* Xs = (lbf16*)dynsmem;
  int r0  = blockIdx.x * 16;
  int tid = threadIdx.x;
  if (use_lds) {
    int kc = K >> 3;                                   // 16B chunks per row
    int row = tid & 15;
    for (int j = tid >> 4; j < kc; j += 16)
      ((lu32x4*)Xs)[row * kc + j] =
          *(const u32x4*)(X + (size_t)(r0 + row) * ldx + j * 8);
  }
  __syncthreads();
  int wave = tid >> 5, lane = tid & 31;
  int c0 = blockIdx.y * 256 + wave * 32;
  if (c0 >= N) return;
  int cb = (lane >> 4) << 3;
  v8f acc0, acc1;
#pragma unroll
  for (int i = 0; i < 8; ++i) {
    acc0[i] = bias[c0 + cb + i];
    acc1[i] = bias[c0 + 16 + cb + i];
  }
  const bf16* Aw0 = WT + (size_t)c0 * ldk;
  const bf16* Aw1 = WT + (size_t)(c0 + 16) * ldk;
  if (use_lds)
    gemm_core((const lbf16*)Xs, K, Aw0, Aw1, ldk, K, lane, acc0, acc1);
  else
    gemm_core(X + (size_t)r0 * ldx, ldx, Aw0, Aw1, ldk, K, lane, acc0, acc1);
  int m = r0 + (lane & 15);
#pragma unroll
  for (int tIdx = 0; tIdx < 2; ++tIdx) {
    v8f acc = tIdx ? acc1 : acc0;
    int nc = c0 + tIdx * 16 + cb;
    if (outb) {
      union { u32x4 q; bf16 h[8]; } o;
#pragma unroll
      for (int i = 0; i < 8; ++i) {
        float v = acc[i];
        if (act) v = v > 0.f ? v : (__expf(v) - 1.f);
        o.h[i] = (bf16)v;
      }
      *(u32x4*)(outb + (size_t)m * ldob + nc) = o.q;
    }
    if (outf) {
#pragma unroll
      for (int i = 0; i < 8; ++i) {
        float v = acc[i];
        if (act) v = v > 0.f ? v : (__expf(v) - 1.f);
        outf[(size_t)m * ldof + nc + i] = v;
      }
    }
  }
}

// ---------------------------------------------------------------------------
// Fused GRU step: 6 gate tiles per wave (x*Wi r/u/n, h*Wh r/u/n) in registers;
// one B fragment feeds 3 WMMAs. Gate math + h update fused; gate matrices
// never hit memory. WiT [3072 x 1056], WhT [3072 x 1024]; x,h staged in LDS.
// ---------------------------------------------------------------------------
__global__ void __launch_bounds__(256) k_gru(
    const bf16* __restrict__ xB, const bf16* __restrict__ hBin,
    const float* __restrict__ hFin, const bf16* __restrict__ WiT,
    const bf16* __restrict__ WhT, const float* __restrict__ bi,
    const float* __restrict__ bh, float* __restrict__ hFout,
    bf16* __restrict__ hBout, bf16* __restrict__ qinB, bf16* __restrict__ latB) {
  extern __shared__ char dynsmem[];
  lbf16* xs = (lbf16*)dynsmem;            // 16 x 1056
  lbf16* hs = xs + 16 * 1056;             // 16 x 1024
  int r0  = blockIdx.x * 16;
  int tid = threadIdx.x;
  int row = tid & 15;
  for (int j = tid >> 4; j < 132; j += 16)
    ((lu32x4*)xs)[row * 132 + j] =
        *(const u32x4*)(xB + (size_t)(r0 + row) * 1056 + j * 8);
  for (int j = tid >> 4; j < 128; j += 16)
    ((lu32x4*)hs)[row * 128 + j] =
        *(const u32x4*)(hBin + (size_t)(r0 + row) * 1024 + j * 8);
  __syncthreads();
  int wave = tid >> 5, lane = tid & 31;
  int c = blockIdx.y * 128 + wave * 16;
  int cb = (lane >> 4) << 3;
  v8f air, aiu, ain, ahr, ahu, ahn;
#pragma unroll
  for (int i = 0; i < 8; ++i) {
    air[i] = bi[c + cb + i];
    aiu[i] = bi[1024 + c + cb + i];
    ain[i] = bi[2048 + c + cb + i];
    ahr[i] = bh[c + cb + i];
    ahu[i] = bh[1024 + c + cb + i];
    ahn[i] = bh[2048 + c + cb + i];
  }
  const bf16* Ar = WiT + (size_t)c * 1056;
  const bf16* Au = WiT + (size_t)(1024 + c) * 1056;
  const bf16* An = WiT + (size_t)(2048 + c) * 1056;
  for (int k = 0; k < 1056; k += 32) {
    v16bf b = load_frag((const lbf16*)xs, 1056, k, lane);
    air = wmma_bf16(load_frag(Ar, 1056, k, lane), b, air);
    aiu = wmma_bf16(load_frag(Au, 1056, k, lane), b, aiu);
    ain = wmma_bf16(load_frag(An, 1056, k, lane), b, ain);
  }
  const bf16* Hr = WhT + (size_t)c * 1024;
  const bf16* Hu = WhT + (size_t)(1024 + c) * 1024;
  const bf16* Hn = WhT + (size_t)(2048 + c) * 1024;
  for (int k = 0; k < 1024; k += 32) {
    v16bf b = load_frag((const lbf16*)hs, 1024, k, lane);
    ahr = wmma_bf16(load_frag(Hr, 1024, k, lane), b, ahr);
    ahu = wmma_bf16(load_frag(Hu, 1024, k, lane), b, ahu);
    ahn = wmma_bf16(load_frag(Hn, 1024, k, lane), b, ahn);
  }
  int m = r0 + (lane & 15);
  int col0 = c + cb;
  union { u32x4 q; bf16 h[8]; } o;
#pragma unroll
  for (int i = 0; i < 8; ++i) {
    float r  = 1.f / (1.f + __expf(-(air[i] + ahr[i])));
    float u  = 1.f / (1.f + __expf(-(aiu[i] + ahu[i])));
    float nn = tanhf(ain[i] + r * ahn[i]);
    float hnew = (1.f - u) * nn + u * hFin[(size_t)m * 1024 + col0 + i];
    hFout[(size_t)m * 1024 + col0 + i] = hnew;
    o.h[i] = (bf16)hnew;
  }
  *(u32x4*)(hBout + (size_t)m * 1024 + col0) = o.q;
  *(u32x4*)(qinB  + (size_t)m * 1536 + col0) = o.q;  // posterior input cols [0,1024)
  *(u32x4*)(latB  + (size_t)m * 2048 + col0) = o.q;  // latent cols [0,1024)
}

// --- per-step small kernels -------------------------------------------------
__global__ void k_prep(const float* __restrict__ act_t,
                       const float* __restrict__ obs_t, bf16* __restrict__ xB,
                       int* __restrict__ idx) {
  int b = threadIdx.x;  // 256
  for (int j = 0; j < 16; ++j)
    xB[b * 1056 + 1024 + j] = (bf16)act_t[b * 16 + j];
  int id = 0;
  for (int j = 0; j < 512; ++j)
    if (obs_t[b * 512 + j] > 0.5f) id = j;   // one-hot argmax
  idx[b] = id;
}

__global__ void k_enc1(const float* __restrict__ We1, const float* __restrict__ be1,
                       const int* __restrict__ idx, bf16* __restrict__ e1) {
  int i = blockIdx.x * 256 + threadIdx.x;  // 256*64
  int b = i >> 6, c = i & 63;
  float v = We1[idx[b] * 64 + c] + be1[c];
  v = v > 0.f ? v : (__expf(v) - 1.f);
  e1[i] = (bf16)v;
}

// softmax/KL/Gumbel-argmax: block = one batch row, wave = one category
__global__ void __launch_bounds__(1024) k_sample(
    const float* __restrict__ postL, const float* __restrict__ priorL,
    const float* __restrict__ gum_t, bf16* __restrict__ xB,
    bf16* __restrict__ latB, float* __restrict__ kl_row, int t) {
  __shared__ float klc[32];
  int b = blockIdx.x, cat = threadIdx.x >> 5, cls = threadIdx.x & 31;
  int off = b * 1024 + cat * 32 + cls;
  float lp = postL[off], lq = priorL[off];
  float mp = wred_max(lp);
  float ep = __expf(lp - mp);
  float sp = wred_sum(ep);
  float prob  = ep / sp;
  float lpost = lp - mp - __logf(sp);
  float mq = wred_max(lq);
  float sq = wred_sum(__expf(lq - mq));
  float lprior = lq - mq - __logf(sq);
  float klsum = wred_sum(prob * (lpost - lprior));  // kl_a == kl_b in value
  float v = lp + gum_t[off];
  int   ai = cls;
#pragma unroll
  for (int m = 16; m > 0; m >>= 1) {
    float ov = __shfl_xor(v, m, 32);
    int   oi = __shfl_xor(ai, m, 32);
    if (ov > v || (ov == v && oi < ai)) { v = ov; ai = oi; }
  }
  float z = (cls == ai) ? 1.f : 0.f;  // straight-through forward value == one-hot
  xB  [b * 1056 + cat * 32 + cls]        = (bf16)z;
  latB[b * 2048 + 1024 + cat * 32 + cls] = (bf16)z;
  if (cls == 0) klc[cat] = klsum;
  __syncthreads();
  if (threadIdx.x == 0) {
    float s = 0.f;
    for (int i = 0; i < 32; ++i) s += klc[i];
    kl_row[t * 256 + b] = s;
  }
}

__global__ void k_ce(const float* __restrict__ logits, const int* __restrict__ idx,
                     float* __restrict__ ce_row, int t) {
  __shared__ float red[256];
  int b = blockIdx.x, tid = threadIdx.x;
  float v0 = logits[b * 512 + tid], v1 = logits[b * 512 + 256 + tid];
  red[tid] = fmaxf(v0, v1);
  __syncthreads();
  for (int s = 128; s > 0; s >>= 1) {
    if (tid < s) red[tid] = fmaxf(red[tid], red[tid + s]);
    __syncthreads();
  }
  float mx = red[0];
  __syncthreads();
  red[tid] = __expf(v0 - mx) + __expf(v1 - mx);
  __syncthreads();
  for (int s = 128; s > 0; s >>= 1) {
    if (tid < s) red[tid] += red[tid + s];
    __syncthreads();
  }
  if (tid == 0)
    ce_row[t * 256 + b] = mx + __logf(red[0]) - logits[b * 512 + idx[b]];
}

__global__ void k_small(const bf16* __restrict__ u, const float* __restrict__ Wr2,
                        const float* __restrict__ br2, const float* __restrict__ Wc2,
                        const float* __restrict__ bc2, const float* __restrict__ rew,
                        const float* __restrict__ done, float* __restrict__ mse_row,
                        float* __restrict__ bce_row, int t) {
  int b = threadIdx.x;  // 256
  float rp = br2[0], cp = bc2[0];
  for (int j = 0; j < 64; ++j) {
    rp += (float)u[b * 192 + 64 + j] * Wr2[j];
    cp += (float)u[b * 192 + 128 + j] * Wc2[j];
  }
  float r  = rew[t * 256 + b];
  float rt = (r > 0.f ? 1.f : (r < 0.f ? -1.f : 0.f)) * log1pf(fabsf(r));
  mse_row[t * 256 + b] = (rp - rt) * (rp - rt);
  float ct = 1.f - done[t * 256 + b];
  bce_row[t * 256 + b] = fmaxf(cp, 0.f) - cp * ct + log1pf(__expf(-fabsf(cp)));
}

__global__ void k_final(const float* __restrict__ ce_row, const float* __restrict__ mse_row,
                        const float* __restrict__ bce_row, const float* __restrict__ kl_row,
                        float* __restrict__ out) {
  __shared__ float sc[64], sm[64], sb[64], sk[64];
  int t = threadIdx.x;  // 64
  float c = 0.f, m = 0.f, bv = 0.f, k = 0.f;
  for (int b = 0; b < 256; ++b) {
    c  += ce_row [t * 256 + b];
    m  += mse_row[t * 256 + b];
    bv += bce_row[t * 256 + b];
    k  += kl_row [t * 256 + b];
  }
  sc[t] = c / 256.f; sm[t] = m / 256.f; sb[t] = bv / 256.f;
  sk[t] = fmaxf(k / 256.f, 1.0f);   // 0.8*max(kl,FB)+0.2*max(kl,FB), kl_a==kl_b
  __syncthreads();
  if (t == 0) {
    float C = 0.f, M = 0.f, Bv = 0.f, K = 0.f;
    for (int i = 0; i < 64; ++i) { C += sc[i]; M += sm[i]; Bv += sb[i]; K += sk[i]; }
    C /= 64.f; M /= 64.f; Bv /= 64.f; K /= 64.f;
    out[0] = C + M + Bv + K; out[1] = C; out[2] = M; out[3] = Bv; out[4] = K;
  }
}

// --- setup kernels ----------------------------------------------------------
// WT[n*ldk + k] = (k < Ksrc) ? W[k*Nsrc + n] : 0   (bf16 transposed weights)
__global__ void k_trans(const float* __restrict__ W, bf16* __restrict__ WT,
                        int Ksrc, int Nsrc, int ldk, int total) {
  int i = blockIdx.x * 256 + threadIdx.x;
  if (i >= total) return;
  int n = i / ldk, k = i % ldk;
  WT[i] = (k < Ksrc) ? (bf16)W[(size_t)k * Nsrc + n] : (bf16)0.f;
}
// WcatT[192 x 2048] from Wd1|Wr1|Wc1 (each [2048 x 64]); bcat[192]
__global__ void k_catT(const float* __restrict__ Wd1, const float* __restrict__ Wr1,
                       const float* __restrict__ Wc1, const float* __restrict__ bd1,
                       const float* __restrict__ br1, const float* __restrict__ bc1,
                       bf16* __restrict__ WcatT, float* __restrict__ bcat) {
  int i = blockIdx.x * 256 + threadIdx.x;
  if (i < 192)
    bcat[i] = i < 64 ? bd1[i] : (i < 128 ? br1[i - 64] : bc1[i - 128]);
  if (i < 192 * 2048) {
    int n = i / 2048, k = i % 2048;
    const float* src = n < 64 ? Wd1 : (n < 128 ? Wr1 : Wc1);
    WcatT[i] = (bf16)src[(size_t)k * 64 + (n & 63)];
  }
}
__global__ void k_zero_bf(bf16* __restrict__ d, int n) {
  int i = blockIdx.x * 256 + threadIdx.x;
  if (i < n) d[i] = (bf16)0.f;
}
__global__ void k_zero_f(float* __restrict__ d, int n) {
  int i = blockIdx.x * 256 + threadIdx.x;
  if (i < n) d[i] = 0.f;
}

// ---------------------------------------------------------------------------
extern "C" void kernel_launch(void* const* d_in, const int* in_sizes, int n_in,
                              void* d_out, int out_size, void* d_ws, size_t ws_size,
                              hipStream_t stream) {
  (void)in_sizes; (void)n_in; (void)out_size; (void)ws_size;
  const float* obs  = (const float*)d_in[0];
  const float* actp = (const float*)d_in[1];
  const float* rew  = (const float*)d_in[2];
  const float* done = (const float*)d_in[3];
  const float* gum  = (const float*)d_in[4];
  const float* We1 = (const float*)d_in[5];  const float* be1 = (const float*)d_in[6];
  const float* We2 = (const float*)d_in[7];  const float* be2 = (const float*)d_in[8];
  const float* Wi  = (const float*)d_in[9];  const float* Wh  = (const float*)d_in[10];
  const float* bi  = (const float*)d_in[11]; const float* bh  = (const float*)d_in[12];
  const float* Wp1 = (const float*)d_in[13]; const float* bp1 = (const float*)d_in[14];
  const float* Wp2 = (const float*)d_in[15]; const float* bp2 = (const float*)d_in[16];
  const float* Wq1 = (const float*)d_in[17]; const float* bq1 = (const float*)d_in[18];
  const float* Wq2 = (const float*)d_in[19]; const float* bq2 = (const float*)d_in[20];
  const float* Wd1 = (const float*)d_in[21]; const float* bd1 = (const float*)d_in[22];
  const float* Wd2 = (const float*)d_in[23]; const float* bd2 = (const float*)d_in[24];
  const float* Wr1 = (const float*)d_in[25]; const float* br1 = (const float*)d_in[26];
  const float* Wr2 = (const float*)d_in[27]; const float* br2 = (const float*)d_in[28];
  const float* Wc1 = (const float*)d_in[29]; const float* bc1 = (const float*)d_in[30];
  const float* Wc2 = (const float*)d_in[31]; const float* bc2 = (const float*)d_in[32];

  char* ws = (char*)d_ws;
  size_t off = 0;
  auto alloc = [&](size_t bytes) -> void* {
    off = (off + 255) & ~(size_t)255;
    void* p = ws + off;
    off += bytes;
    return p;
  };
  bf16* WiT   = (bf16*)alloc((size_t)3072 * 1056 * 2);
  bf16* WhT   = (bf16*)alloc((size_t)3072 * 1024 * 2);
  bf16* Wp1T  = (bf16*)alloc((size_t)1024 * 1024 * 2);
  bf16* Wp2T  = (bf16*)alloc((size_t)1024 * 1024 * 2);
  bf16* Wq1T  = (bf16*)alloc((size_t)1024 * 1536 * 2);
  bf16* Wq2T  = (bf16*)alloc((size_t)1024 * 1024 * 2);
  bf16* We2T  = (bf16*)alloc((size_t)512 * 64 * 2);
  bf16* Wd2T  = (bf16*)alloc((size_t)512 * 64 * 2);
  bf16* WcatT = (bf16*)alloc((size_t)192 * 2048 * 2);
  float* bcat = (float*)alloc(192 * 4);
  bf16* xB    = (bf16*)alloc((size_t)256 * 1056 * 2);
  bf16* hB0   = (bf16*)alloc((size_t)256 * 1024 * 2);
  bf16* hB1   = (bf16*)alloc((size_t)256 * 1024 * 2);
  float* hF0  = (float*)alloc((size_t)256 * 1024 * 4);
  float* hF1  = (float*)alloc((size_t)256 * 1024 * 4);
  bf16* qinB  = (bf16*)alloc((size_t)256 * 1536 * 2);
  bf16* p1B   = (bf16*)alloc((size_t)256 * 1024 * 2);
  bf16* q1B   = (bf16*)alloc((size_t)256 * 1024 * 2);
  float* priorL = (float*)alloc((size_t)256 * 1024 * 4);
  float* postL  = (float*)alloc((size_t)256 * 1024 * 4);
  bf16* latB  = (bf16*)alloc((size_t)256 * 2048 * 2);
  bf16* uB    = (bf16*)alloc((size_t)256 * 192 * 2);
  bf16* e1B   = (bf16*)alloc((size_t)256 * 64 * 2);
  float* reconL = (float*)alloc((size_t)256 * 512 * 4);
  float* ce_row  = (float*)alloc((size_t)64 * 256 * 4);
  float* mse_row = (float*)alloc((size_t)64 * 256 * 4);
  float* bce_row = (float*)alloc((size_t)64 * 256 * 4);
  float* kl_row  = (float*)alloc((size_t)64 * 256 * 4);
  int* idxb = (int*)alloc(256 * 4);

  auto TR = [&](const float* W, bf16* WT, int Ksrc, int Nsrc, int ldk) {
    int total = Nsrc * ldk;
    k_trans<<<dim3((total + 255) / 256), dim3(256), 0, stream>>>(W, WT, Ksrc,
                                                                 Nsrc, ldk, total);
  };
  TR(Wi, WiT, 1040, 3072, 1056);   // K padded 1040 -> 1056 (zeros)
  TR(Wh, WhT, 1024, 3072, 1024);
  TR(Wp1, Wp1T, 1024, 1024, 1024);
  TR(Wp2, Wp2T, 1024, 1024, 1024);
  TR(Wq1, Wq1T, 1536, 1024, 1536);
  TR(Wq2, Wq2T, 1024, 1024, 1024);
  TR(We2, We2T, 64, 512, 64);
  TR(Wd2, Wd2T, 64, 512, 64);
  k_catT<<<dim3((192 * 2048 + 255) / 256), dim3(256), 0, stream>>>(
      Wd1, Wr1, Wc1, bd1, br1, bc1, WcatT, bcat);
  k_zero_bf<<<dim3((256 * 1056 + 255) / 256), dim3(256), 0, stream>>>(xB, 256 * 1056);
  k_zero_bf<<<dim3((256 * 1024 + 255) / 256), dim3(256), 0, stream>>>(hB0, 256 * 1024);
  k_zero_f <<<dim3((256 * 1024 + 255) / 256), dim3(256), 0, stream>>>(hF0, 256 * 1024);

  for (int t = 0; t < 64; ++t) {
    const float* act_t = actp + (size_t)t * 256 * 16;
    const float* obs_t = obs  + (size_t)t * 256 * 512;
    const float* gum_t = gum  + (size_t)t * 256 * 1024;
    const float* hFin = (t & 1) ? hF1 : hF0;
    float*       hFout = (t & 1) ? hF0 : hF1;
    const bf16*  hBin = (t & 1) ? hB1 : hB0;
    bf16*        hBout = (t & 1) ? hB0 : hB1;

    k_prep<<<dim3(1), dim3(256), 0, stream>>>(act_t, obs_t, xB, idxb);
    k_enc1<<<dim3(64), dim3(256), 0, stream>>>(We1, be1, idxb, e1B);
    // embed = e1 @ We2 + be2 -> qin cols [1024,1536)
    k_gemm<<<dim3(16, 2), dim3(256), 16 * 64 * 2, stream>>>(
        e1B, 64, We2T, 64, be2, 64, 512, qinB + 1024, 1536, nullptr, 0, 0, 1);
    k_gru<<<dim3(16, 8), dim3(256), (16 * 1056 + 16 * 1024) * 2, stream>>>(
        xB, hBin, hFin, WiT, WhT, bi, bh, hFout, hBout, qinB, latB);
    // prior/post hidden layers (ELU)
    k_gemm<<<dim3(16, 4), dim3(256), 16 * 1024 * 2, stream>>>(
        hBout, 1024, Wp1T, 1024, bp1, 1024, 1024, p1B, 1024, nullptr, 0, 1, 1);
    k_gemm<<<dim3(16, 4), dim3(256), 16 * 1536 * 2, stream>>>(
        qinB, 1536, Wq1T, 1536, bq1, 1536, 1024, q1B, 1024, nullptr, 0, 1, 1);
    // logits (fp32)
    k_gemm<<<dim3(16, 4), dim3(256), 16 * 1024 * 2, stream>>>(
        p1B, 1024, Wp2T, 1024, bp2, 1024, 1024, nullptr, 0, priorL, 1024, 0, 1);
    k_gemm<<<dim3(16, 4), dim3(256), 16 * 1024 * 2, stream>>>(
        q1B, 1024, Wq2T, 1024, bq2, 1024, 1024, nullptr, 0, postL, 1024, 0, 1);
    k_sample<<<dim3(256), dim3(1024), 0, stream>>>(
        postL, priorL, gum_t, xB, latB, kl_row, t);
    // fused decoder/reward/continue first layers: latent @ [Wd1|Wr1|Wc1]
    k_gemm<<<dim3(16, 1), dim3(256), 0, stream>>>(
        latB, 2048, WcatT, 2048, bcat, 2048, 192, uB, 192, nullptr, 0, 1, 0);
    // recon logits = u[:, :64] @ Wd2 + bd2
    k_gemm<<<dim3(16, 2), dim3(256), 16 * 64 * 2, stream>>>(
        uB, 192, Wd2T, 64, bd2, 64, 512, nullptr, 0, reconL, 512, 0, 1);
    k_ce<<<dim3(256), dim3(256), 0, stream>>>(reconL, idxb, ce_row, t);
    k_small<<<dim3(1), dim3(256), 0, stream>>>(
        uB, Wr2, br2, Wc2, bc2, rew, done, mse_row, bce_row, t);
  }
  k_final<<<dim3(1), dim3(64), 0, stream>>>(ce_row, mse_row, bce_row, kl_row,
                                            (float*)d_out);
}
